// Attention_4836133175904
// MI455X (gfx1250) — compile-verified, matrix operand order checked by reading, not run on hardware
//
#include <hip/hip_runtime.h>

// Multi-head causal attention, MI455X (gfx1250, wave32, WMMA).
// B=2, T=2048, E=1024, H=16, D=64. All matmuls use v_wmma_f32_16x16x32_bf16
// (fp32 accumulate). Flash-attention keeps the T x T scores on-chip
// (materializing them = ~1GB HBM traffic, ~45us at 23.3 TB/s -- dominates).
// This revision: 32-row tiles (B fragments amortized over 2 M-subtiles:
// 6 loads / 8 WMMAs instead of 10 / 4) and explicit ping-pong prefetch so
// WMMAs overlap the next k-step's loads instead of stalling on
// s_wait_loadcnt 0. V fragments are loaded before the softmax VALU block so
// the exp/reduction latency hides the V traffic.
// Workspace: ~50 MB of d_ws for bf16 x/weights/Q/K/V^T/att.

#define BB 2
#define TT 2048
#define EE 1024
#define HH 16
#define DD 64

typedef __attribute__((ext_vector_type(16))) __bf16 v16bf;
typedef __attribute__((ext_vector_type(8)))  __bf16 v8bf;
typedef __attribute__((ext_vector_type(8)))  float  v8f;

// Load a 16x32 bf16 A/B fragment from a [16, K] row-major tile (ld = row
// stride in elements). Wave32 layout (ISA 7.12.2): lane l holds row (l&15),
// K chunks [h*8, h*8+8) and [16+h*8, 16+h*8+8) where h = l>>4.
__device__ __forceinline__ v16bf load_frag_b16(const __bf16* p, int ld) {
  int lane = threadIdx.x & 31;
  int row  = lane & 15;
  int half = lane >> 4;
  const __bf16* q = p + row * ld + half * 8;
  v8bf lo = *(const v8bf*)(q);
  v8bf hi = *(const v8bf*)(q + 16);
  return __builtin_shufflevector(lo, hi, 0,1,2,3,4,5,6,7,8,9,10,11,12,13,14,15);
}

__device__ __forceinline__ v8f wmma_bf16(v16bf a, v16bf b, v8f c) {
  // (neg_a, A, neg_b, B, c_mod, C, reuse_a, reuse_b)
  return __builtin_amdgcn_wmma_f32_16x16x32_bf16(false, a, false, b, (short)0, c,
                                                 false, false);
}

__global__ void cast_f32_bf16(const float* __restrict__ in, __bf16* __restrict__ out, int n) {
  int i = blockIdx.x * blockDim.x + threadIdx.x;
  if (i < n) out[i] = (__bf16)in[i];
}

// Wq/Wk/Wv: [H, E, D] f32 -> [H, D, E] bf16 (GEMM B-operand gets K=E contiguous)
__global__ void transpose_w(const float* __restrict__ in, __bf16* __restrict__ out) {
  int i = blockIdx.x * blockDim.x + threadIdx.x;
  if (i >= HH * EE * DD) return;
  int h = i / (EE * DD);
  int r = i % (EE * DD);
  int e = r / DD;
  int d = r % DD;
  out[(h * DD + d) * EE + e] = (__bf16)in[i];
}

// One wave computes a 32x64 tile of Q, K, or V^T for one (h, b).
// grid = (T/32, H*B, 3), block = 32. Software-pipelined k-loop.
__global__ void qkv_proj(const __bf16* __restrict__ xb,
                         const __bf16* __restrict__ Wqb,
                         const __bf16* __restrict__ Wkb,
                         const __bf16* __restrict__ Wvb,
                         __bf16* __restrict__ Q,
                         __bf16* __restrict__ Km,
                         __bf16* __restrict__ Vt) {
  int t0 = blockIdx.x * 32;
  int hb = blockIdx.y;              // h * B + b
  int c  = blockIdx.z;              // 0 = Q, 1 = K, 2 = V^T
  int h  = hb / BB, b = hb % BB;
  const __bf16* W = (c == 0 ? Wqb : (c == 1 ? Wkb : Wvb)) + (size_t)h * DD * EE;
  const __bf16* X = xb + ((size_t)b * TT + t0) * EE;

  v8f acc[2][4] = {};
  // Prime the pipeline: fragments for k0 = 0
  v16bf a0 = load_frag_b16(X, EE);
  v16bf a1 = load_frag_b16(X + 16 * EE, EE);
  v16bf bb[4];
#pragma unroll
  for (int nt = 0; nt < 4; ++nt) bb[nt] = load_frag_b16(W + nt * 16 * EE, EE);

#pragma unroll 2
  for (int k0 = 0; k0 < EE; k0 += 32) {
    int kn = (k0 + 32 < EE) ? (k0 + 32) : 0;   // clamped: last prefetch is harmless
    v16bf na0 = load_frag_b16(X + kn, EE);
    v16bf na1 = load_frag_b16(X + 16 * EE + kn, EE);
    v16bf nb[4];
#pragma unroll
    for (int nt = 0; nt < 4; ++nt) nb[nt] = load_frag_b16(W + nt * 16 * EE + kn, EE);
#pragma unroll
    for (int nt = 0; nt < 4; ++nt) {
      acc[0][nt] = wmma_bf16(a0, bb[nt], acc[0][nt]);
      acc[1][nt] = wmma_bf16(a1, bb[nt], acc[1][nt]);
    }
    a0 = na0; a1 = na1;
#pragma unroll
    for (int nt = 0; nt < 4; ++nt) bb[nt] = nb[nt];
  }

  int lane = threadIdx.x & 31;
  int n = lane & 15, half = lane >> 4;
  if (c < 2) {
    __bf16* dst = (c == 0 ? Q : Km) + (size_t)hb * TT * DD;
#pragma unroll
    for (int rt = 0; rt < 2; ++rt)
#pragma unroll
      for (int nt = 0; nt < 4; ++nt)
#pragma unroll
        for (int r = 0; r < 8; ++r)
          dst[(t0 + rt * 16 + r + 8 * half) * DD + nt * 16 + n] = (__bf16)acc[rt][nt][r];
  } else {
    __bf16* dst = Vt + (size_t)hb * DD * TT;   // V^T: [D, T], T contiguous
#pragma unroll
    for (int rt = 0; rt < 2; ++rt)
#pragma unroll
      for (int nt = 0; nt < 4; ++nt)
#pragma unroll
        for (int r = 0; r < 8; ++r)
          dst[(nt * 16 + n) * TT + t0 + rt * 16 + r + 8 * half] = (__bf16)acc[rt][nt][r];
  }
}

// Flash attention: one wave per (h, b, 32-row Q tile). Online softmax, causal.
// grid = (T/32, H*B), block = 32.
__global__ void flash_attn(const __bf16* __restrict__ Q,
                           const __bf16* __restrict__ Km,
                           const __bf16* __restrict__ Vt,
                           __bf16* __restrict__ att) {
  int t0 = blockIdx.x * 32;
  int hb = blockIdx.y;
  int h  = hb / BB, b = hb % BB;
  const __bf16* Qb = Q  + ((size_t)hb * TT + t0) * DD;
  const __bf16* Kb = Km + (size_t)hb * TT * DD;
  const __bf16* Vb = Vt + (size_t)hb * DD * TT;

  int lane = threadIdx.x & 31;
  int n16 = lane & 15, half = lane >> 4;

  // Q fragments: 2 row-subtiles x 2 k-steps (K-dim = D = 64)
  v16bf qa[2][2];
#pragma unroll
  for (int rt = 0; rt < 2; ++rt) {
    qa[rt][0] = load_frag_b16(Qb + rt * 16 * DD + 0,  DD);
    qa[rt][1] = load_frag_b16(Qb + rt * 16 * DD + 32, DD);
  }

  v8f o[2][4] = {};
  float Mr[16], Lr[16];
#pragma unroll
  for (int i = 0; i < 16; ++i) { Mr[i] = -1e30f; Lr[i] = 0.f; }

  __shared__ __attribute__((aligned(16))) __bf16 pl[2 * 16 * 32];  // P staging
  const float inv_scale = 0.02209708691207961f;  // 1/sqrt(T): reference scales by sqrt(seq_len)

  for (int j0 = 0; j0 <= t0 + 31; j0 += 32) {
    // ---- S = Q (32x64) @ K^T (64x32): batch the 4 K fragments, then 8 WMMAs
    v16bf kb[4];
    kb[0] = load_frag_b16(Kb + (j0 + 0)  * DD + 0,  DD);
    kb[1] = load_frag_b16(Kb + (j0 + 0)  * DD + 32, DD);
    kb[2] = load_frag_b16(Kb + (j0 + 16) * DD + 0,  DD);
    kb[3] = load_frag_b16(Kb + (j0 + 16) * DD + 32, DD);
    v8f s[2][2] = {};
#pragma unroll
    for (int rt = 0; rt < 2; ++rt) {
      s[rt][0] = wmma_bf16(qa[rt][0], kb[0], s[rt][0]);
      s[rt][0] = wmma_bf16(qa[rt][1], kb[1], s[rt][0]);
      s[rt][1] = wmma_bf16(qa[rt][0], kb[2], s[rt][1]);
      s[rt][1] = wmma_bf16(qa[rt][1], kb[3], s[rt][1]);
    }

    // Issue V fragment loads now: their latency overlaps the softmax VALU work.
    v16bf vb[4];
#pragma unroll
    for (int nt = 0; nt < 4; ++nt)
      vb[nt] = load_frag_b16(Vb + (nt * 16) * TT + j0, TT);

    // ---- Online softmax. C layout: VGPR r = row (r + 8*half), column n16.
#pragma unroll
    for (int rt = 0; rt < 2; ++rt) {
#pragma unroll
      for (int r = 0; r < 8; ++r) {
        int tq = t0 + rt * 16 + r + 8 * half;
        float v0 = (j0 + n16      <= tq) ? s[rt][0][r] * inv_scale : -1e30f;
        float v1 = (j0 + 16 + n16 <= tq) ? s[rt][1][r] * inv_scale : -1e30f;
        float rowmax = fmaxf(v0, v1);
#pragma unroll
        for (int off = 1; off < 16; off <<= 1)
          rowmax = fmaxf(rowmax, __shfl_xor(rowmax, off, 32));
        float newM  = fmaxf(Mr[rt * 8 + r], rowmax);
        float alpha = __expf(Mr[rt * 8 + r] - newM);
        Mr[rt * 8 + r] = newM;
        float p0 = __expf(v0 - newM);
        float p1 = __expf(v1 - newM);
        pl[rt * 512 + (r + 8 * half) * 32 +      n16] = (__bf16)p0;
        pl[rt * 512 + (r + 8 * half) * 32 + 16 + n16] = (__bf16)p1;
        float psum = p0 + p1;
#pragma unroll
        for (int off = 1; off < 16; off <<= 1)
          psum += __shfl_xor(psum, off, 32);
        Lr[rt * 8 + r] = Lr[rt * 8 + r] * alpha + psum;
#pragma unroll
        for (int nt = 0; nt < 4; ++nt) o[rt][nt][r] *= alpha;
      }
    }
    __syncthreads();  // single wave: the required LDS wait before re-reading pl

    // ---- O += P (32x32) @ V (32x64); V^T rows are the B-operand [N=d, K=key]
    v16bf pa0 = load_frag_b16(pl, 32);
    v16bf pa1 = load_frag_b16(pl + 512, 32);
#pragma unroll
    for (int nt = 0; nt < 4; ++nt) {
      o[0][nt] = wmma_bf16(pa0, vb[nt], o[0][nt]);
      o[1][nt] = wmma_bf16(pa1, vb[nt], o[1][nt]);
    }
    __syncthreads();  // before the next iteration overwrites pl
  }

  // Normalize and write att[b, t, h*D + d] in bf16
#pragma unroll
  for (int rt = 0; rt < 2; ++rt)
#pragma unroll
    for (int r = 0; r < 8; ++r) {
      float li = 1.f / Lr[rt * 8 + r];
      int trow = t0 + rt * 16 + r + 8 * half;
#pragma unroll
      for (int nt = 0; nt < 4; ++nt)
        att[((size_t)b * TT + trow) * EE + h * DD + nt * 16 + n16] =
            (__bf16)(o[rt][nt][r] * li);
    }
}

// out = att @ Wo^T. Wo row-major [Eout, Ein] is already the [N, K] B-operand.
// One wave per 32x64 output tile; software-pipelined. Output fp32.
// grid = (B*T/32, E/64), block = 32.
__global__ void out_proj(const __bf16* __restrict__ att,
                         const __bf16* __restrict__ Wob,
                         float* __restrict__ out) {
  int m0 = blockIdx.x * 32;
  int n0 = blockIdx.y * 64;
  const __bf16* A  = att + (size_t)m0 * EE;
  const __bf16* Bm = Wob + (size_t)n0 * EE;

  v8f acc[2][4] = {};
  v16bf a0 = load_frag_b16(A, EE);
  v16bf a1 = load_frag_b16(A + 16 * EE, EE);
  v16bf bb[4];
#pragma unroll
  for (int nt = 0; nt < 4; ++nt) bb[nt] = load_frag_b16(Bm + nt * 16 * EE, EE);

#pragma unroll 2
  for (int k0 = 0; k0 < EE; k0 += 32) {
    int kn = (k0 + 32 < EE) ? (k0 + 32) : 0;
    v16bf na0 = load_frag_b16(A + kn, EE);
    v16bf na1 = load_frag_b16(A + 16 * EE + kn, EE);
    v16bf nb[4];
#pragma unroll
    for (int nt = 0; nt < 4; ++nt) nb[nt] = load_frag_b16(Bm + nt * 16 * EE + kn, EE);
#pragma unroll
    for (int nt = 0; nt < 4; ++nt) {
      acc[0][nt] = wmma_bf16(a0, bb[nt], acc[0][nt]);
      acc[1][nt] = wmma_bf16(a1, bb[nt], acc[1][nt]);
    }
    a0 = na0; a1 = na1;
#pragma unroll
    for (int nt = 0; nt < 4; ++nt) bb[nt] = nb[nt];
  }

  int lane = threadIdx.x & 31;
  int n = lane & 15, half = lane >> 4;
#pragma unroll
  for (int rt = 0; rt < 2; ++rt)
#pragma unroll
    for (int nt = 0; nt < 4; ++nt)
#pragma unroll
      for (int r = 0; r < 8; ++r)
        out[(size_t)(m0 + rt * 16 + r + 8 * half) * EE + n0 + nt * 16 + n] =
            acc[rt][nt][r];
}

extern "C" void kernel_launch(void* const* d_in, const int* in_sizes, int n_in,
                              void* d_out, int out_size, void* d_ws, size_t ws_size,
                              hipStream_t stream) {
  (void)in_sizes; (void)n_in; (void)out_size; (void)ws_size;
  const float* x  = (const float*)d_in[0];
  // d_in[1] = boolean causal mask: applied analytically (tk > tq), not read.
  const float* Wq = (const float*)d_in[2];
  const float* Wk = (const float*)d_in[3];
  const float* Wv = (const float*)d_in[4];
  const float* Wo = (const float*)d_in[5];
  float* out = (float*)d_out;

  // Workspace layout (bf16); total ~50.3 MB
  char* wsb = (char*)d_ws;
  size_t off = 0;
  __bf16* xb   = (__bf16*)(wsb + off); off += (size_t)BB * TT * EE * 2;
  __bf16* Wqb  = (__bf16*)(wsb + off); off += (size_t)HH * DD * EE * 2;
  __bf16* Wkb  = (__bf16*)(wsb + off); off += (size_t)HH * DD * EE * 2;
  __bf16* Wvb  = (__bf16*)(wsb + off); off += (size_t)HH * DD * EE * 2;
  __bf16* Wob  = (__bf16*)(wsb + off); off += (size_t)EE * EE * 2;
  __bf16* Qs   = (__bf16*)(wsb + off); off += (size_t)HH * BB * TT * DD * 2;
  __bf16* Ks   = (__bf16*)(wsb + off); off += (size_t)HH * BB * TT * DD * 2;
  __bf16* Vts  = (__bf16*)(wsb + off); off += (size_t)HH * BB * TT * DD * 2;
  __bf16* atts = (__bf16*)(wsb + off); off += (size_t)BB * TT * EE * 2;

  const int nx = BB * TT * EE;
  cast_f32_bf16<<<(nx + 255) / 256, 256, 0, stream>>>(x, xb, nx);
  const int nw = HH * EE * DD;
  transpose_w<<<(nw + 255) / 256, 256, 0, stream>>>(Wq, Wqb);
  transpose_w<<<(nw + 255) / 256, 256, 0, stream>>>(Wk, Wkb);
  transpose_w<<<(nw + 255) / 256, 256, 0, stream>>>(Wv, Wvb);
  const int nwo = EE * EE;
  cast_f32_bf16<<<(nwo + 255) / 256, 256, 0, stream>>>(Wo, Wob, nwo);

  qkv_proj<<<dim3(TT / 32, HH * BB, 3), 32, 0, stream>>>(xb, Wqb, Wkb, Wvb, Qs, Ks, Vts);
  flash_attn<<<dim3(TT / 32, HH * BB), 32, 0, stream>>>(Qs, Ks, Vts, atts);
  out_proj<<<dim3(BB * TT / 32, EE / 64), 32, 0, stream>>>(atts, Wob, out);
}
